// R_SCNN_17918603559427
// MI455X (gfx1250) — matrix operand order repaired and checked by reading, not compiled, not run on hardware
//
#include <hip/hip_runtime.h>
#include <hip/hip_bf16.h>

typedef __attribute__((ext_vector_type(16))) __bf16 v16bf;
typedef __attribute__((ext_vector_type(8)))  __bf16 v8bf;
typedef __attribute__((ext_vector_type(8)))  float  v8f;

constexpr int kC  = 256;   // channels
constexpr int kH  = 64;    // height
constexpr int kW  = 128;   // width (scan axis)
constexpr int kB  = 2;     // batch
constexpr int kHT = 16;    // h-rows per workgroup (one WMMA N-tile)
constexpr int kBUF = 2 * kHT * kC;   // one carry buffer (r plane + s plane)

static __device__ __forceinline__ v8f wmma_bf16(v16bf a, v16bf b, v8f c) {
    // D = A(16x32 bf16) * B(32x16 bf16) + C(16x16 f32)
    return __builtin_amdgcn_wmma_f32_16x16x32_bf16(
        /*neg_a=*/false, a, /*neg_b=*/false, b,
        /*c_mod=*/(short)0, c, /*reuse_a=*/false, /*reuse_b=*/false);
}

// One workgroup per (batch, h-tile); 256 threads = 8 waves; wave i owns output
// channel tiles ot = 2i, 2i+1. Both directional SCNN passes fused:
//   r_w = c_w + relu(M r_{w-1});  s_w = r_w + relu(M s_{w-1})
// Loop-invariant data (M fragments, conv1 weights) is VGPR-resident for the
// whole scan. Carries are double-buffered in LDS, transposed [h][c] bf16 so
// B-operands are single contiguous 32B reads -> exactly ONE barrier per step.
// Column loads use uniform-base (SADDR) + invariant VGPR offset + immediate
// per-o offsets, so the step only advances one scalar pointer.
__global__ void __launch_bounds__(256, 1)
scnn_scan_kernel(const float* __restrict__ p2,
                 const float* __restrict__ convw,
                 const float* __restrict__ conv1w,
                 float* __restrict__ ybuf,   // [B][H][W] channel-reduced map
                 float* __restrict__ sums) { // [2], zeroed here for kernel 2
    extern __shared__ char smem[];
    __bf16* carr = (__bf16*)smem;            // [2 buf][2 (r,s)][16][256] = 32 KB

    const int tid   = threadIdx.x;
    const int lane  = tid & 31;
    const int lid   = lane & 15;     // row-within-tile / N index
    const int hi    = lane >> 4;     // half-wave select per ISA operand layout
    const int wave  = tid >> 5;      // 0..7
    const int b     = blockIdx.x >> 2;
    const int htile = blockIdx.x & 3;
    const int hglob = htile * kHT + lid;

    if (blockIdx.x == 0 && tid < 2) sums[tid] = 0.f;

    // Zero this block's slice of ybuf (accumulated via global atomics below).
    for (int i = tid; i < kHT * kW; i += 256)
        ybuf[(b * kH + htile * kHT + (i >> 7)) * kW + (i & (kW - 1))] = 0.f;

    // Loop-invariant VGPR-resident operands:
    //  - A-fragments of M = conv_w[:,:,0,4]. A in this code's contiguous k order: lanes<16
    //    hold K {c0..c0+15}, lanes>=16 hold K {c0+16..31} (matches B),
    //    row M = lid.
    //  - c1r: the 16 conv1 weights matching this lane's D-fragment rows.
    v16bf Areg[2][8];
    v8f   c1r[2];
#pragma unroll
    for (int t = 0; t < 2; ++t) {
        const int o = (wave * 2 + t) * 16 + lid;
        const float* mrow = convw + (size_t)o * kC * 9 + 4;  // stride 9 per c
#pragma unroll
        for (int ck = 0; ck < 8; ++ck) {
            const int c0 = ck * 32;
#pragma unroll
            for (int e = 0; e < 8; ++e) {
                Areg[t][ck][e]     = (__bf16)mrow[(c0 + hi * 16 + e) * 9];
                Areg[t][ck][8 + e] = (__bf16)mrow[(c0 + hi * 16 + 8 + e) * 9];
            }
        }
        const int obase = (wave * 2 + t) * 16 + hi * 8;
#pragma unroll
        for (int j = 0; j < 8; ++j) c1r[t][j] = conv1w[obase + j];
    }

    // Column 0 into buffer 0: r_0 = s_0 = c_0; its reduction -> output col W-1.
    // (ybuf slice zeroing above is by this same block; fence before atomics.)
    __syncthreads();
    for (int i = tid; i < kC * kHT; i += 256) {
        const int c = i >> 4, hh = i & 15;
        const float v = p2[((b * kC + c) * kH + htile * kHT + hh) * kW + 0];
        const __bf16 bv = (__bf16)v;
        carr[hh * kC + c]            = bv;   // r plane, buf 0
        carr[kHT * kC + hh * kC + c] = bv;   // s plane, buf 0
        atomicAdd(&ybuf[(b * kH + htile * kHT + hh) * kW + (kW - 1)],
                  conv1w[c] * v);
    }
    __syncthreads();

    // Uniform (scalar) base advancing 4B/step; per-lane invariant offset in a
    // single VGPR; per-o parts become 24-bit immediate offsets (max 8.36 MB).
    const float* puni = p2 + (size_t)b * kC * kH * kW + 1;   // at column w
    const int    loff = hglob * kW;                          // lane-invariant

    for (int w = 1; w < kW; ++w) {
        // One 128B line covers 32 consecutive w at fixed (o,h): prefetch ahead.
        if ((w & 31) == 1) {
            int dp = 31; if (w + dp > kW - 1) dp = kW - 1 - w;
#pragma unroll
            for (int t = 0; t < 2; ++t)
#pragma unroll
                for (int j = 0; j < 8; ++j) {
                    const int o = (wave * 2 + t) * 16 + hi * 8 + j;
                    __builtin_prefetch(&puni[loff + o * (kH * kW) + dp], 0, 3);
                }
        }

        // Ping-pong carry buffers: read pin, write pout; ONE barrier per step.
        const __bf16* Bin  = carr + ((w - 1) & 1) * kBUF;
        __bf16*       Bout = carr + (w & 1) * kBUF;
        // B in the same contiguous k order as A: lanes<16 hold K=c0..c0+15 at N=lid,
        // lanes>=16 hold K=c0+16..31 -> one contiguous 32B read from [h][c].
        const __bf16* BrowR = Bin + lid * kC + hi * 16;
        const __bf16* BrowS = BrowR + kHT * kC;

        v8f accR[2], accS[2];
#pragma unroll
        for (int t = 0; t < 2; ++t)
#pragma unroll
            for (int e = 0; e < 8; ++e) { accR[t][e] = 0.f; accS[t][e] = 0.f; }

#pragma unroll
        for (int ck = 0; ck < 8; ++ck) {
            const int c0 = ck * 32;
            const v16bf br = *(const v16bf*)(BrowR + c0);
            const v16bf bs = *(const v16bf*)(BrowS + c0);
            accR[0] = wmma_bf16(Areg[0][ck], br, accR[0]);
            accR[1] = wmma_bf16(Areg[1][ck], br, accR[1]);
            accS[0] = wmma_bf16(Areg[0][ck], bs, accS[0]);
            accS[1] = wmma_bf16(Areg[1][ck], bs, accS[1]);
        }

        // D layout: VGPR j holds (o = ot*16 + hi*8 + j, h = ht*16 + lid).
        float yp = 0.f;
#pragma unroll
        for (int t = 0; t < 2; ++t) {
            const int obase = (wave * 2 + t) * 16 + hi * 8;
            v8bf nrv, nsv;
#pragma unroll
            for (int j = 0; j < 8; ++j) {
                const int o = obase + j;
                const float col = puni[loff + o * (kH * kW)];  // saddr + imm
                const float nr = col + fmaxf(accR[t][j], 0.f);
                const float ns = nr + fmaxf(accS[t][j], 0.f);
                nrv[j] = (__bf16)nr;
                nsv[j] = (__bf16)ns;
                yp += c1r[t][j] * ns;              // weights are VGPR-resident
            }
            *(v8bf*)(Bout + lid * kC + obase)            = nrv;  // packed 16B
            *(v8bf*)(Bout + kHT * kC + lid * kC + obase) = nsv;
        }
        // lanes L and L+16 share h: combine, then one global atomic per h/wave.
        yp += __shfl_xor(yp, 16);
        if (hi == 0)
            atomicAdd(&ybuf[(b * kH + hglob) * kW + (kW - 1 - w)], yp);

        __syncthreads();   // the single per-step fence (double-buffered carries)
        puni += 1;         // scalar pointer advance (SALU, off the VALU path)
    }
}

// Bilinear x4 (align_corners) on the already channel-reduced map, storing the
// raw value to d_out and wave-reducing sum/sumsq for the single-channel BN.
__global__ void upsample_sum_kernel(const float* __restrict__ y,
                                    float* __restrict__ out,
                                    float* __restrict__ sums) {
    const int idx = blockIdx.x * blockDim.x + threadIdx.x;  // 2*256*512
    const int b   = idx >> 17;
    const int rem = idx & ((1 << 17) - 1);
    const int ho  = rem >> 9;
    const int wo  = rem & 511;
    const float ysf = ho * (63.0f / 255.0f);
    const float xsf = wo * (127.0f / 511.0f);
    const int y0 = (int)ysf, x0 = (int)xsf;
    const float wy = ysf - (float)y0, wx = xsf - (float)x0;
    const int y1 = (y0 + 1 > 63) ? 63 : y0 + 1;
    const int x1 = (x0 + 1 > 127) ? 127 : x0 + 1;
    const float* yb = y + b * kH * kW;
    const float v00 = yb[y0 * kW + x0], v01 = yb[y0 * kW + x1];
    const float v10 = yb[y1 * kW + x0], v11 = yb[y1 * kW + x1];
    float v = (v00 + (v01 - v00) * wx) * (1.f - wy)
            + (v10 + (v11 - v10) * wx) * wy;
    out[idx] = v;
    float v2 = v * v;
#pragma unroll
    for (int off = 16; off > 0; off >>= 1) {   // wave32 shuffle reduce
        v  += __shfl_down(v,  off);
        v2 += __shfl_down(v2, off);
    }
    if ((threadIdx.x & 31) == 0) {
        atomicAdd(&sums[0], v);
        atomicAdd(&sums[1], v2);
    }
}

__global__ void bn_sigmoid_kernel(float* __restrict__ out,
                                  const float* __restrict__ sums,
                                  const float* __restrict__ gamma,
                                  const float* __restrict__ beta) {
    const int idx = blockIdx.x * blockDim.x + threadIdx.x;
    const float n = (float)(kB * 256 * 512);
    const float mean = sums[0] / n;
    const float var  = sums[1] / n - mean * mean;
    const float inv  = rsqrtf(var + 1e-5f);
    const float x = (out[idx] - mean) * inv * gamma[0] + beta[0];
    out[idx] = 1.f / (1.f + __expf(-x));
}

extern "C" void kernel_launch(void* const* d_in, const int* in_sizes, int n_in,
                              void* d_out, int out_size, void* d_ws, size_t ws_size,
                              hipStream_t stream) {
    (void)in_sizes; (void)n_in; (void)out_size; (void)ws_size;
    const float* p2     = (const float*)d_in[0];
    const float* convw  = (const float*)d_in[1];
    const float* conv1w = (const float*)d_in[2];
    const float* gamma  = (const float*)d_in[3];
    const float* beta   = (const float*)d_in[4];
    float* out  = (float*)d_out;
    float* ybuf = (float*)d_ws;                 // kB*kH*kW f32 = 64 KB
    float* sums = ybuf + kB * kH * kW;          // 2 f32

    const size_t smem = (size_t)(2 * kBUF) * sizeof(__bf16);   // 32 KB

    scnn_scan_kernel<<<kB * 4, 256, smem, stream>>>(p2, convw, conv1w, ybuf, sums);

    const int total = kB * 256 * 512;           // 262144
    upsample_sum_kernel<<<total / 256, 256, 0, stream>>>(ybuf, out, sums);
    bn_sigmoid_kernel<<<total / 256, 256, 0, stream>>>(out, sums, gamma, beta);
}
